// GptOssAttention_52536039965401
// MI455X (gfx1250) — matrix operand order repaired and checked by reading, not compiled, not run on hardware
//
#include <hip/hip_runtime.h>
#include <hip/hip_bf16.h>

typedef __attribute__((ext_vector_type(16))) __bf16 v16bf;
typedef __attribute__((ext_vector_type(8)))  float  v8f;

union FragA {
    v16bf v;
    uint4 q[2];
};

#define T_LEN 2048
#define HIDDEN_DIM 2880
#define NUM_HEADS 64
#define NUM_KV_HEADS 8
#define HEAD_DIM 64
#define Q_SIZE 4096
#define KV_SIZE 512
#define QKV_COLS 5120
#define WINDOW 128
#define SCALE_F 0.125f
#define MASKV (-1.0e30f)

// ---------------------------------------------------------------------------
// f32 -> bf16 convert
// ---------------------------------------------------------------------------
__global__ void cvt_bf16_kernel(const float* __restrict__ src,
                                __bf16* __restrict__ dst, int n) {
    int i = blockIdx.x * blockDim.x + threadIdx.x;
    if (i < n) dst[i] = (__bf16)src[i];
}

// ---------------------------------------------------------------------------
// GEMM: C[M,N] f32 = A[M,K] bf16 (row major) * B[K,N] bf16 (row major)
// Block tile 128 x BN, BK=32, 256 threads (8 waves, 4M x 2N).
// Wave tile 32 x (BN/2): 2 x (BN/32) WMMA accumulators.
//   BN=128 -> 8 WMMA / wave / k-step (QKV proj, N=5120)
//   BN=64  -> 4 WMMA / wave / k-step (O proj,   N=2880)
// ---------------------------------------------------------------------------
#define BM 128
#define BK 32

template <int BN>
__global__ __launch_bounds__(256) void gemm_bf16_kernel(
    const __bf16* __restrict__ A, const __bf16* __restrict__ B,
    float* __restrict__ C, int M, int N, int K) {
    constexpr int NI = BN / 32;  // 16-wide column tiles per wave

    __shared__ __align__(16) __bf16 As[BM * BK];  // [m][k], k stride 32
    __shared__ __align__(16) __bf16 Bs[BN * BK];  // transposed: [n][k]

    const int tid  = threadIdx.x;
    const int wave = tid >> 5;
    const int lane = tid & 31;
    const int half = lane >> 4;
    const int l16  = lane & 15;
    const int waveM = wave & 3;   // 0..3 -> 32-row strip
    const int waveN = wave >> 2;  // 0..1 -> (BN/2)-col strip

    const int m0 = blockIdx.y * BM;
    const int n0 = blockIdx.x * BN;

    v8f acc[2][NI] = {};

    for (int k0 = 0; k0 < K; k0 += BK) {
        // stage A tile 128x32 (each thread: one half-row, 16 bf16 = 2x b128)
        {
            int row = tid >> 1;
            int ch  = (tid & 1) * 16;
            const uint4* src =
                reinterpret_cast<const uint4*>(A + (size_t)(m0 + row) * K + k0 + ch);
            uint4* dst = reinterpret_cast<uint4*>(&As[row * BK + ch]);
            dst[0] = src[0];
            dst[1] = src[1];
        }
        // stage B tile 32xBN, transposed into Bs[n][k]
        {
            constexpr int LOADS = BN / 64;  // uint4 loads per thread
#pragma unroll
            for (int u = 0; u < LOADS; ++u) {
                int flat = tid * LOADS + u;        // 0 .. 4*BN-1
                int krow = flat / (BN / 8);
                int nn   = (flat % (BN / 8)) * 8;
                uint4 d = *reinterpret_cast<const uint4*>(
                    B + (size_t)(k0 + krow) * N + n0 + nn);
                const __bf16* e = reinterpret_cast<const __bf16*>(&d);
#pragma unroll
                for (int i = 0; i < 8; ++i) Bs[(nn + i) * BK + krow] = e[i];
            }
        }
        __syncthreads();

        FragA a[2], b[NI];
#pragma unroll
        for (int mi = 0; mi < 2; ++mi) {
            int r = waveM * 32 + mi * 16 + l16;
            const uint4* p = reinterpret_cast<const uint4*>(&As[r * BK]);
            a[mi].q[0] = p[half];      // K = half*8 .. +7
            a[mi].q[1] = p[2 + half];  // K = 16+half*8 .. +7
        }
#pragma unroll
        for (int ni = 0; ni < NI; ++ni) {
            int c = waveN * (BN / 2) + ni * 16 + l16;
            const uint4* p = reinterpret_cast<const uint4*>(&Bs[c * BK]);
            b[ni].q[0] = p[half];      // K = half*8 .. +7
            b[ni].q[1] = p[2 + half];  // K = 16+half*8 .. +7
        }
#pragma unroll
        for (int mi = 0; mi < 2; ++mi)
#pragma unroll
            for (int ni = 0; ni < NI; ++ni)
                acc[mi][ni] = __builtin_amdgcn_wmma_f32_16x16x32_bf16(
                    false, a[mi].v, false, b[ni].v, (short)0, acc[mi][ni],
                    false, false);
        __syncthreads();
    }

#pragma unroll
    for (int mi = 0; mi < 2; ++mi) {
        int rbase = m0 + waveM * 32 + mi * 16 + half * 8;
#pragma unroll
        for (int ni = 0; ni < NI; ++ni) {
            int cidx = n0 + waveN * (BN / 2) + ni * 16 + l16;
#pragma unroll
            for (int g = 0; g < 8; ++g)
                C[(size_t)(rbase + g) * N + cidx] = acc[mi][ni][g];
        }
    }
}

// ---------------------------------------------------------------------------
// RoPE (NeoX) on Q; also folds in the 1/sqrt(HEAD_DIM) scale. Output bf16.
// ---------------------------------------------------------------------------
__global__ void rope_q_kernel(const float* __restrict__ qkv,
                              const int* __restrict__ pos,
                              __bf16* __restrict__ qb) {
    int idx = blockIdx.x * blockDim.x + threadIdx.x;  // T*64*32
    if (idx >= T_LEN * NUM_HEADS * 32) return;
    int i = idx & 31;
    int h = (idx >> 5) & 63;
    int t = idx >> 11;
    float p   = (float)pos[t];
    float inv = powf(150000.0f, -(float)i / 32.0f);
    float fr  = p * inv;
    float c = cosf(fr), s = sinf(fr);
    const float* src = qkv + (size_t)t * QKV_COLS + h * 64;
    float x1 = src[i], x2 = src[i + 32];
    __bf16* dst = qb + (size_t)t * Q_SIZE + h * 64;
    dst[i]      = (__bf16)((x1 * c - x2 * s) * SCALE_F);
    dst[i + 32] = (__bf16)((x2 * c + x1 * s) * SCALE_F);
}

__global__ void rope_k_kernel(const float* __restrict__ qkv,
                              const int* __restrict__ pos,
                              __bf16* __restrict__ kb) {
    int idx = blockIdx.x * blockDim.x + threadIdx.x;  // T*8*32
    if (idx >= T_LEN * NUM_KV_HEADS * 32) return;
    int i = idx & 31;
    int h = (idx >> 5) & 7;
    int t = idx >> 8;
    float p   = (float)pos[t];
    float inv = powf(150000.0f, -(float)i / 32.0f);
    float fr  = p * inv;
    float c = cosf(fr), s = sinf(fr);
    const float* src = qkv + (size_t)t * QKV_COLS + Q_SIZE + h * 64;
    float x1 = src[i], x2 = src[i + 32];
    __bf16* dst = kb + (size_t)t * KV_SIZE + h * 64;
    dst[i]      = (__bf16)(x1 * c - x2 * s);
    dst[i + 32] = (__bf16)(x2 * c + x1 * s);
}

// V transposed to [KV_SIZE, T] bf16 so PV B-fragments are contiguous loads
__global__ void vtr_kernel(const float* __restrict__ qkv,
                           __bf16* __restrict__ vT) {
    int idx = blockIdx.x * blockDim.x + threadIdx.x;  // T*512
    if (idx >= T_LEN * KV_SIZE) return;
    int d = idx & 511;
    int t = idx >> 9;
    vT[(size_t)d * T_LEN + t] =
        (__bf16)qkv[(size_t)t * QKV_COLS + Q_SIZE + KV_SIZE + d];
}

// ---------------------------------------------------------------------------
// Sliding-window flash attention with sinks.
// One wave per (head, 16-query tile). Block = 4 waves. grid = (32, 64).
// ---------------------------------------------------------------------------
__global__ __launch_bounds__(128) void attn_kernel(
    const __bf16* __restrict__ qb,   // [T, 4096] (pre-scaled, roped)
    const __bf16* __restrict__ kb,   // [T, 512]  (roped)
    const __bf16* __restrict__ vT,   // [512, T]
    const float* __restrict__ sinks, // [64]
    __bf16* __restrict__ ob) {       // [T, 4096]
    __shared__ __align__(16) __bf16 P[4][16 * 32];

    const int tid  = threadIdx.x;
    const int w    = tid >> 5;
    const int lane = tid & 31;
    const int half = lane >> 4;
    const int l16  = lane & 15;
    const int h    = blockIdx.y;
    const int kvh  = h >> 3;
    const int qt   = blockIdx.x * 4 + w;  // 0..127
    const int q0   = qt * 16;

    // zero P once (cols 16..31 stay zero forever -> zero-padded A fragments)
    for (int i = tid; i < 4 * 16 * 32; i += 128)
        reinterpret_cast<__bf16*>(P)[i] = (__bf16)0.0f;
    __syncthreads();

    // Q fragments: two 16x32 A tiles (dims 0-31, 32-63)
    FragA a0, a1;
    {
        const uint4* p = reinterpret_cast<const uint4*>(
            qb + (size_t)(q0 + l16) * Q_SIZE + h * 64);
        a0.q[0] = p[half];
        a0.q[1] = p[2 + half];
        a1.q[0] = p[4 + half];
        a1.q[1] = p[6 + half];
    }

    float m[8], l[8];
    v8f o[4] = {};
#pragma unroll
    for (int g = 0; g < 8; ++g) { m[g] = -50.0f; l[g] = 0.0f; }

    for (int j = 0; j < 9; ++j) {
        int kt = qt - 8 + j;
        bool active = (kt >= 0);
        if (active) {
            // K^T fragments: B[d][key], lane = key column, contiguous dim loads
            FragA b0, b1;
            const uint4* p = reinterpret_cast<const uint4*>(
                kb + (size_t)(kt * 16 + l16) * KV_SIZE + kvh * 64);
            b0.q[0] = p[half];
            b0.q[1] = p[2 + half];
            b1.q[0] = p[4 + half];
            b1.q[1] = p[6 + half];

            v8f s = {};
            s = __builtin_amdgcn_wmma_f32_16x16x32_bf16(false, a0.v, false, b0.v,
                                                        (short)0, s, false, false);
            s = __builtin_amdgcn_wmma_f32_16x16x32_bf16(false, a1.v, false, b1.v,
                                                        (short)0, s, false, false);
            int col = kt * 16 + l16;
#pragma unroll
            for (int g = 0; g < 8; ++g) {
                int row = q0 + g + half * 8;
                float v = s[g];
                bool ok = (col <= row) && (row - col < WINDOW);
                v = ok ? v : MASKV;
                float rm = v;  // row max across the 16 lanes of this half
                rm = fmaxf(rm, __shfl_xor(rm, 1, 32));
                rm = fmaxf(rm, __shfl_xor(rm, 2, 32));
                rm = fmaxf(rm, __shfl_xor(rm, 4, 32));
                rm = fmaxf(rm, __shfl_xor(rm, 8, 32));
                float mn = fmaxf(m[g], rm);
                float sf = expf(m[g] - mn);
                float pe = expf(v - mn);  // masked -> 0
                float rs = pe;
                rs += __shfl_xor(rs, 1, 32);
                rs += __shfl_xor(rs, 2, 32);
                rs += __shfl_xor(rs, 4, 32);
                rs += __shfl_xor(rs, 8, 32);
                l[g] = l[g] * sf + rs;
                m[g] = mn;
#pragma unroll
                for (int c = 0; c < 4; ++c) o[c][g] *= sf;
                P[w][(g + half * 8) * 32 + l16] = (__bf16)pe;
            }
        }
        __syncthreads();  // uniform: P stores visible for fragment reload
        if (active) {
            FragA ap;
            const uint4* pp =
                reinterpret_cast<const uint4*>(&P[w][l16 * 32]);
            ap.q[0] = pp[half * 2];
            ap.q[1] = pp[half * 2 + 1];  // half=1: zeros (cols 16..31)

            int kcol = kt * 16 + half * 16;            // half=1 lanes don't-care
            if (kcol > T_LEN - 16) kcol = T_LEN - 16;  // keep reads in-bounds
#pragma unroll
            for (int c = 0; c < 4; ++c) {
                FragA bv;
                const uint4* vp = reinterpret_cast<const uint4*>(
                    vT + (size_t)(kvh * 64 + c * 16 + l16) * T_LEN + kcol);
                bv.q[0] = vp[0];
                bv.q[1] = vp[1];
                o[c] = __builtin_amdgcn_wmma_f32_16x16x32_bf16(
                    false, ap.v, false, bv.v, (short)0, o[c], false, false);
            }
        }
        __syncthreads();
    }

    float snk = sinks[h];
#pragma unroll
    for (int g = 0; g < 8; ++g) {
        float mt = fmaxf(m[g], snk);
        float sf = expf(m[g] - mt);
        float denom = l[g] * sf + expf(snk - mt);
        float r = sf / denom;
        int row = q0 + g + half * 8;
#pragma unroll
        for (int c = 0; c < 4; ++c)
            ob[(size_t)row * Q_SIZE + h * 64 + c * 16 + l16] =
                (__bf16)(o[c][g] * r);
    }
}

// ---------------------------------------------------------------------------
// Launch
// ---------------------------------------------------------------------------
extern "C" void kernel_launch(void* const* d_in, const int* in_sizes, int n_in,
                              void* d_out, int out_size, void* d_ws,
                              size_t ws_size, hipStream_t stream) {
    const int*   positions = (const int*)d_in[0];
    const float* hidden    = (const float*)d_in[1];
    const float* w_qkv     = (const float*)d_in[2];
    const float* w_o       = (const float*)d_in[3];
    const float* sinks     = (const float*)d_in[4];
    float*       out       = (float*)d_out;

    char* ws = (char*)d_ws;
    size_t off = 0;
    auto alloc = [&](size_t bytes) -> char* {
        char* p = ws + off;
        off = (off + bytes + 255) & ~(size_t)255;
        return p;
    };
    __bf16* hb   = (__bf16*)alloc((size_t)T_LEN * HIDDEN_DIM * 2);
    __bf16* wqb  = (__bf16*)alloc((size_t)HIDDEN_DIM * QKV_COLS * 2);
    __bf16* wob  = (__bf16*)alloc((size_t)Q_SIZE * HIDDEN_DIM * 2);
    float*  qkvf = (float*)alloc((size_t)T_LEN * QKV_COLS * 4);
    __bf16* qb2  = (__bf16*)alloc((size_t)T_LEN * Q_SIZE * 2);
    __bf16* kb2  = (__bf16*)alloc((size_t)T_LEN * KV_SIZE * 2);
    __bf16* vtb  = (__bf16*)alloc((size_t)KV_SIZE * T_LEN * 2 + 256);
    __bf16* ab   = (__bf16*)alloc((size_t)T_LEN * Q_SIZE * 2);

    int n1 = T_LEN * HIDDEN_DIM;
    cvt_bf16_kernel<<<(n1 + 255) / 256, 256, 0, stream>>>(hidden, hb, n1);
    int n2 = HIDDEN_DIM * QKV_COLS;
    cvt_bf16_kernel<<<(n2 + 255) / 256, 256, 0, stream>>>(w_qkv, wqb, n2);
    int n3 = Q_SIZE * HIDDEN_DIM;
    cvt_bf16_kernel<<<(n3 + 255) / 256, 256, 0, stream>>>(w_o, wob, n3);

    // QKV projection: [2048,2880] x [2880,5120] -> f32 (BN=128: 5120/128=40)
    gemm_bf16_kernel<128><<<dim3(QKV_COLS / 128, T_LEN / BM), 256, 0, stream>>>(
        hb, wqb, qkvf, T_LEN, QKV_COLS, HIDDEN_DIM);

    int nq = T_LEN * NUM_HEADS * 32;
    rope_q_kernel<<<(nq + 255) / 256, 256, 0, stream>>>(qkvf, positions, qb2);
    int nk = T_LEN * NUM_KV_HEADS * 32;
    rope_k_kernel<<<(nk + 255) / 256, 256, 0, stream>>>(qkvf, positions, kb2);
    int nv = T_LEN * KV_SIZE;
    vtr_kernel<<<(nv + 255) / 256, 256, 0, stream>>>(qkvf, vtb);

    attn_kernel<<<dim3(T_LEN / 16 / 4, NUM_HEADS), 128, 0, stream>>>(
        qb2, kb2, vtb, sinks, ab);

    // Output projection: [2048,4096] x [4096,2880] -> d_out f32 (BN=64: 2880/64=45)
    gemm_bf16_kernel<64><<<dim3(HIDDEN_DIM / 64, T_LEN / BM), 256, 0, stream>>>(
        ab, wob, out, T_LEN, HIDDEN_DIM, Q_SIZE);
}
